// MultiHeadAttention_77498389889173
// MI455X (gfx1250) — matrix operand order repaired and checked by reading, not compiled, not run on hardware
//
#include <hip/hip_runtime.h>
#include <hip/hip_bf16.h>
#include <math.h>

typedef __bf16 bf16;
typedef __attribute__((ext_vector_type(16))) __bf16 v16bf;
typedef __attribute__((ext_vector_type(4)))  __bf16 v4bf;
typedef __attribute__((ext_vector_type(8)))  float  v8f;

#define BATCH 4
#define SEQ   2048
#define DM    1024
#define NH    16
#define HD    64
#define NROWS (BATCH * SEQ)   // 8192
#define LDP   72              // padded LDS K-stride (144 B, 16B aligned)

static __device__ inline v8f v8f_zero() {
  v8f z = {0.f, 0.f, 0.f, 0.f, 0.f, 0.f, 0.f, 0.f};
  return z;
}

static __device__ inline v8f wmma_bf16(v16bf a, v16bf b, v8f c) {
  return __builtin_amdgcn_wmma_f32_16x16x32_bf16(false, a, false, b, (short)0, c,
                                                 false, false);
}

static __device__ inline unsigned lds_addr_of(const void* p) {
  return (unsigned)(uintptr_t)p;  // low 32 bits of generic ptr == LDS offset
}

static __device__ inline void async_copy_b128(unsigned lds, const void* gptr) {
  asm volatile("global_load_async_to_lds_b128 %0, %1, off"
               :: "v"(lds), "v"((unsigned long long)(uintptr_t)gptr)
               : "memory");
}

static __device__ inline void wait_async0() {
  asm volatile("s_wait_asynccnt 0" ::: "memory");
}
static __device__ inline void wait_ds0() {
  asm volatile("s_wait_dscnt 0" ::: "memory");
}

// A fragment (16x32 bf16, row-major source, stride ld).  Lane L: row m =
// row0+(L&15); holds k = kb..kb+15, kb = 16*(L>>4).
static __device__ inline v16bf load_a_bf16(const bf16* src, int ld, int row0,
                                           int k0, int lane) {
  const int m = row0 + (lane & 15);
  const bf16* p = src + (size_t)m * ld + k0 + ((lane >> 4) << 4);
  v16bf a;
#pragma unroll
  for (int i = 0; i < 8; ++i) a[i] = p[i];
#pragma unroll
  for (int i = 0; i < 8; ++i) a[8 + i] = p[8 + i];
  return a;
}

// B fragment (32x16 bf16) from an N-major source (rows = N, cols = K
// contiguous): lane L holds column n = n0+(L&15), k = kb..kb+15, kb=16*(L>>4).
static __device__ inline v16bf load_bt_bf16(const bf16* src, int ld, int n0,
                                            int k0, int lane) {
  const int n = n0 + (lane & 15);
  const bf16* p = src + (size_t)n * ld + k0 + ((lane >> 4) << 4);
  v16bf b;
#pragma unroll
  for (int i = 0; i < 16; ++i) b[i] = p[i];
  return b;
}

// B fragment (32x16 bf16) from a K-major (row-major K x N) source.
static __device__ inline v16bf load_b_bf16(const bf16* src, int ld, int k0,
                                           int n0, int lane) {
  const int n = n0 + (lane & 15);
  const int kb = k0 + ((lane >> 4) << 4);
  v16bf b;
#pragma unroll
  for (int i = 0; i < 16; ++i) b[i] = src[(size_t)(kb + i) * ld + n];
  return b;
}

// Stage a 64x256 fp32 W tile transposed into WsT[256][LDP] as bf16, packing
// 4 consecutive K per N into b64 DS stores.  256 threads.
static __device__ inline void stage_w_t(const float* __restrict__ W, int k0,
                                        int col0, bf16 (*WsT)[LDP], int tid) {
  const int kq = (tid & 15) * 4;   // 4 consecutive K rows
  const int ng = (tid >> 4) * 16;  // 16 N columns
#pragma unroll
  for (int c = 0; c < 4; ++c) {
    float w[4][4];
#pragma unroll
    for (int i = 0; i < 4; ++i)
#pragma unroll
      for (int j = 0; j < 4; ++j)
        w[i][j] = W[(size_t)(k0 + kq + i) * DM + col0 + ng + 4 * c + j];
#pragma unroll
    for (int j = 0; j < 4; ++j) {
      v4bf t = {(bf16)w[0][j], (bf16)w[1][j], (bf16)w[2][j], (bf16)w[3][j]};
      *(v4bf*)&WsT[ng + 4 * c + j][kq] = t;
    }
  }
}

// ---------------------------------------------------------------------------
// Kernel 1: fused QKV projection.  Y = X @ W (fp32 in, bf16 out) in
// [B, H, S, HD] layout.  Tile 64x256, K-step 64.
// grid = (DM/256, NROWS/64, 3), block = 256 (8 waves, each 32x64).
// ---------------------------------------------------------------------------
__global__ __launch_bounds__(256) void qkv_proj_kernel(
    const float* __restrict__ X, const float* __restrict__ Wq,
    const float* __restrict__ Wk, const float* __restrict__ Wv,
    bf16* __restrict__ qb, bf16* __restrict__ kb, bf16* __restrict__ vb) {
  const float* W = (blockIdx.z == 0) ? Wq : (blockIdx.z == 1) ? Wk : Wv;
  bf16* Y = (blockIdx.z == 0) ? qb : (blockIdx.z == 1) ? kb : vb;

  __shared__ bf16 Xs[64][LDP];    // 64 rows x 64 K
  __shared__ bf16 WsT[256][LDP];  // 256 N x 64 K (transposed)

  const int tid = threadIdx.x, lane = tid & 31, wid = tid >> 5;
  const int row0 = blockIdx.y * 64, col0 = blockIdx.x * 256;
  const int wr = wid & 1, wc = wid >> 1;

  v8f acc[2][4];
#pragma unroll
  for (int mt = 0; mt < 2; ++mt)
#pragma unroll
    for (int nt = 0; nt < 4; ++nt) acc[mt][nt] = v8f_zero();

  const int xr = tid >> 2, xc = (tid & 3) * 16;

  for (int k0 = 0; k0 < DM; k0 += 64) {
    const float* xp = X + (size_t)(row0 + xr) * DM + k0 + xc;
#pragma unroll
    for (int i = 0; i < 16; ++i) Xs[xr][xc + i] = (bf16)xp[i];
    stage_w_t(W, k0, col0, WsT, tid);
    __syncthreads();

#pragma unroll
    for (int kc = 0; kc < 2; ++kc) {
      v16bf af[2];
#pragma unroll
      for (int mt = 0; mt < 2; ++mt)
        af[mt] = load_a_bf16(&Xs[0][0], LDP, 32 * wr + 16 * mt, 32 * kc, lane);
#pragma unroll
      for (int nt = 0; nt < 4; ++nt) {
        v16bf bw =
            load_bt_bf16(&WsT[0][0], LDP, 64 * wc + 16 * nt, 32 * kc, lane);
        acc[0][nt] = wmma_bf16(af[0], bw, acc[0][nt]);
        acc[1][nt] = wmma_bf16(af[1], bw, acc[1][nt]);
      }
    }
    __syncthreads();
  }

  const int hb8 = (lane >> 4) << 3, n15 = lane & 15;
#pragma unroll
  for (int mt = 0; mt < 2; ++mt)
#pragma unroll
    for (int nt = 0; nt < 4; ++nt)
#pragma unroll
      for (int r = 0; r < 8; ++r) {
        const int R = row0 + 32 * wr + 16 * mt + r + hb8;
        const int C = col0 + 64 * wc + 16 * nt + n15;
        const int b = R >> 11, s = R & (SEQ - 1);
        const int h = C >> 6, hd = C & (HD - 1);
        Y[(((size_t)(b * NH + h)) * SEQ + s) * HD + hd] = (bf16)acc[mt][nt][r];
      }
}

// ---------------------------------------------------------------------------
// Kernel 2: causal flash attention.  One wave owns a 32-row query tile, KV
// stepped in 32s with online softmax.  V tiles are DMA'd into a per-wave LDS
// slice with global_load_async_to_lds_b128, overlapped with the QK^T WMMAs.
// grid = (SEQ/32/8, B*NH), block = 256 (8 waves).
// ---------------------------------------------------------------------------
__global__ __launch_bounds__(256) void attention_kernel(
    const bf16* __restrict__ qb, const bf16* __restrict__ kbuf,
    const bf16* __restrict__ vb, bf16* __restrict__ ctx) {
  __shared__ bf16 Plds[8][32][40];  // per-wave P transpose staging
  __shared__ bf16 Vs[8][32][LDP];   // per-wave async V tile

  const int tid = threadIdx.x, lane = tid & 31, wid = tid >> 5;
  const int bh = blockIdx.y, b = bh >> 4, h = bh & 15;
  const int qbase = (blockIdx.x * 8 + wid) * 32;

  const bf16* Qbh = qb + (size_t)bh * SEQ * HD;
  const bf16* Kbh = kbuf + (size_t)bh * SEQ * HD;
  const bf16* Vbh = vb + (size_t)bh * SEQ * HD;

  const int hb8 = (lane >> 4) << 3, n15 = lane & 15;
  const unsigned vs_base = lds_addr_of(&Vs[wid][0][0]);

  v16bf aq[2][2];
#pragma unroll
  for (int mt = 0; mt < 2; ++mt)
#pragma unroll
    for (int kc = 0; kc < 2; ++kc)
      aq[mt][kc] = load_a_bf16(Qbh, HD, qbase + 16 * mt, 32 * kc, lane);

  v8f octx[2][4];
  float Mrow[2][8], Lrow[2][8];
#pragma unroll
  for (int mt = 0; mt < 2; ++mt) {
#pragma unroll
    for (int cn = 0; cn < 4; ++cn) octx[mt][cn] = v8f_zero();
#pragma unroll
    for (int r = 0; r < 8; ++r) { Mrow[mt][r] = -1e30f; Lrow[mt][r] = 0.f; }
  }

  for (int kv0 = 0; kv0 < qbase + 32; kv0 += 32) {
    // Make sure last iteration's DS reads of Vs are drained, then kick off the
    // async DMA of this step's 32x64 V tile so it overlaps the QK^T WMMAs.
    wait_ds0();
#pragma unroll
    for (int i = 0; i < 8; ++i) {
      const int id = i * 32 + lane;
      const int r = id >> 3, c = id & 7;  // row 0..31, 16B chunk 0..7
      async_copy_b128(vs_base + r * (LDP * 2) + c * 16,
                      Vbh + (size_t)(kv0 + r) * HD + c * 8);
    }
    __builtin_prefetch(Kbh + (size_t)(kv0 + 32) * HD, 0, 1);

    // ---- scores S = Q @ K^T for a 32x32 tile ----
    v8f sc[2][2];
#pragma unroll
    for (int mt = 0; mt < 2; ++mt)
#pragma unroll
      for (int nt = 0; nt < 2; ++nt) sc[mt][nt] = v8f_zero();
#pragma unroll
    for (int nt = 0; nt < 2; ++nt)
#pragma unroll
      for (int kc = 0; kc < 2; ++kc) {
        v16bf bk = load_bt_bf16(Kbh, HD, kv0 + 16 * nt, 32 * kc, lane);
        sc[0][nt] = wmma_bf16(aq[0][kc], bk, sc[0][nt]);
        sc[1][nt] = wmma_bf16(aq[1][kc], bk, sc[1][nt]);
      }

    // ---- scale + causal mask (element-wise, no divergence) ----
#pragma unroll
    for (int mt = 0; mt < 2; ++mt)
#pragma unroll
      for (int nt = 0; nt < 2; ++nt)
#pragma unroll
        for (int r = 0; r < 8; ++r) {
          const int mg = qbase + 16 * mt + r + hb8;
          const int ng = kv0 + 16 * nt + n15;
          const float x = sc[mt][nt][r] * 0.125f;  // 1/sqrt(64)
          sc[mt][nt][r] = (ng > mg) ? -1e30f : x;
        }

    // ---- online softmax update + stage P to LDS (D-layout -> row-major) ----
#pragma unroll
    for (int mt = 0; mt < 2; ++mt) {
#pragma unroll
      for (int r = 0; r < 8; ++r) {
        float mx = fmaxf(sc[mt][0][r], sc[mt][1][r]);
#pragma unroll
        for (int msk = 1; msk <= 8; msk <<= 1)
          mx = fmaxf(mx, __shfl_xor(mx, msk, 32));
        const float mn = fmaxf(Mrow[mt][r], mx);
        const float al = __expf(Mrow[mt][r] - mn);
        const float p0 = __expf(sc[mt][0][r] - mn);
        const float p1 = __expf(sc[mt][1][r] - mn);
        float rs = p0 + p1;
#pragma unroll
        for (int msk = 1; msk <= 8; msk <<= 1) rs += __shfl_xor(rs, msk, 32);
        Lrow[mt][r] = Lrow[mt][r] * al + rs;
        Mrow[mt][r] = mn;
        const int row = 16 * mt + r + hb8;
        Plds[wid][row][n15] = (bf16)p0;
        Plds[wid][row][16 + n15] = (bf16)p1;
#pragma unroll
        for (int cn = 0; cn < 4; ++cn) octx[mt][cn][r] *= al;
      }
    }
    wait_ds0();     // P staged
    wait_async0();  // V tile landed in LDS

    // ---- ctx += P @ V ----
    v16bf ap0 = load_a_bf16(&Plds[wid][0][0], 40, 0, 0, lane);
    v16bf ap1 = load_a_bf16(&Plds[wid][0][0], 40, 16, 0, lane);
#pragma unroll
    for (int cn = 0; cn < 4; ++cn) {
      v16bf bv = load_b_bf16(&Vs[wid][0][0], LDP, 0, 16 * cn, lane);
      octx[0][cn] = wmma_bf16(ap0, bv, octx[0][cn]);
      octx[1][cn] = wmma_bf16(ap1, bv, octx[1][cn]);
    }
  }

  // ---- normalize and write ctx in [B, S, H*HD] layout ----
#pragma unroll
  for (int mt = 0; mt < 2; ++mt)
#pragma unroll
    for (int r = 0; r < 8; ++r) {
      const float inv = 1.0f / Lrow[mt][r];
      const int s = qbase + 16 * mt + r + hb8;
#pragma unroll
      for (int cn = 0; cn < 4; ++cn) {
        const int c = 16 * cn + n15;
        ctx[((size_t)b * SEQ + s) * DM + h * HD + c] =
            (bf16)(octx[mt][cn][r] * inv);
      }
    }
}

// ---------------------------------------------------------------------------
// Kernel 3: output projection.  out = ctx @ Wo + bo (bf16 in, fp32 out).
// ctx tile is DMA'd straight into LDS (already bf16) via async b128 copies.
// grid = (DM/256, NROWS/64), block = 256.
// ---------------------------------------------------------------------------
__global__ __launch_bounds__(256) void out_proj_kernel(
    const bf16* __restrict__ Xc, const float* __restrict__ Wo,
    const float* __restrict__ bo, float* __restrict__ out) {
  __shared__ bf16 Xs[64][LDP];
  __shared__ bf16 WsT[256][LDP];

  const int tid = threadIdx.x, lane = tid & 31, wid = tid >> 5;
  const int row0 = blockIdx.y * 64, col0 = blockIdx.x * 256;
  const int wr = wid & 1, wc = wid >> 1;
  const unsigned xs_base = lds_addr_of(&Xs[0][0]);

  v8f acc[2][4];
#pragma unroll
  for (int mt = 0; mt < 2; ++mt)
#pragma unroll
    for (int nt = 0; nt < 4; ++nt) acc[mt][nt] = v8f_zero();

  for (int k0 = 0; k0 < DM; k0 += 64) {
    wait_ds0();  // previous iteration's Xs reads drained before DMA overwrite
#pragma unroll
    for (int i = 0; i < 2; ++i) {
      const int id = i * 256 + tid;
      const int r = id >> 3, c = id & 7;  // row 0..63, 16B chunk 0..7
      async_copy_b128(xs_base + r * (LDP * 2) + c * 16,
                      Xc + (size_t)(row0 + r) * DM + k0 + c * 8);
    }
    stage_w_t(Wo, k0, col0, WsT, tid);
    wait_async0();
    __syncthreads();

#pragma unroll
    for (int kc = 0; kc < 2; ++kc) {
      v16bf af[2];
#pragma unroll
      for (int mt = 0; mt < 2; ++mt)
        af[mt] = load_a_bf16(&Xs[0][0], LDP, 32 * wr + 16 * mt, 32 * kc, lane);
#pragma unroll
      for (int nt = 0; nt < 4; ++nt) {
        v16bf bw =
            load_bt_bf16(&WsT[0][0], LDP, 64 * wc + 16 * nt, 32 * kc, lane);
        acc[0][nt] = wmma_bf16(af[0], bw, acc[0][nt]);
        acc[1][nt] = wmma_bf16(af[1], bw, acc[1][nt]);
      }
    }
    __syncthreads();
  }

  const int hb8 = (lane >> 4) << 3, n15 = lane & 15;
#pragma unroll
  for (int mt = 0; mt < 2; ++mt)
#pragma unroll
    for (int nt = 0; nt < 4; ++nt)
#pragma unroll
      for (int r = 0; r < 8; ++r) {
        const int R = row0 + 32 * wr + 16 * mt + r + hb8;
        const int C = col0 + 64 * wc + 16 * nt + n15;
        out[(size_t)R * DM + C] = acc[mt][nt][r] + bo[C];
      }
}

// ---------------------------------------------------------------------------
extern "C" void kernel_launch(void* const* d_in, const int* in_sizes, int n_in,
                              void* d_out, int out_size, void* d_ws,
                              size_t ws_size, hipStream_t stream) {
  (void)in_sizes; (void)n_in; (void)out_size; (void)ws_size;
  const float* X  = (const float*)d_in[0];
  const float* Wq = (const float*)d_in[1];
  const float* Wk = (const float*)d_in[2];
  const float* Wv = (const float*)d_in[3];
  const float* Wo = (const float*)d_in[4];
  const float* bo = (const float*)d_in[5];
  float* out = (float*)d_out;

  const size_t elems = (size_t)NROWS * DM;  // 8M elements per tensor
  bf16* qb = (bf16*)d_ws;                   // 16 MB each
  bf16* kb = qb + elems;
  bf16* vb = kb + elems;
  bf16* cb = vb + elems;                    // total 64 MB of workspace

  dim3 blk(256);
  qkv_proj_kernel<<<dim3(DM / 256, NROWS / 64, 3), blk, 0, stream>>>(
      X, Wq, Wk, Wv, qb, kb, vb);
  attention_kernel<<<dim3(SEQ / 32 / 8, BATCH * NH), blk, 0, stream>>>(
      qb, kb, vb, cb);
  out_proj_kernel<<<dim3(DM / 256, NROWS / 64), blk, 0, stream>>>(
      cb, Wo, bo, out);
}
